// Fern_11536282157534
// MI455X (gfx1250) — matrix operand order repaired and checked by reading, not compiled, not run on hardware
//
#include <hip/hip_runtime.h>

// ---------------------------------------------------------------------------
// Fused patch-MLP for MI455X (gfx1250, wave32, WMMA).
// D = A(16x32 bf16) x B(32x16 bf16) + C(16x16 f32) via v_wmma_f32_16x16x32_bf16
// ---------------------------------------------------------------------------

typedef __attribute__((ext_vector_type(16))) __bf16        v16bf;
typedef __attribute__((ext_vector_type(8)))  float         v8f;
typedef __attribute__((ext_vector_type(4)))  unsigned int  u32x4;

#define IMS     64
#define NPS     61                      // N = IMSIZE - PSIZE
#define M_ROWS  (NPS * NPS * 64)        // 238144 rows
#define M_TILE  128
#define NUM_TILES ((M_ROWS + M_TILE - 1) / M_TILE)   // 1861
#define WID     256
#define LDS_T   264                     // bf16 element stride for activation tiles

// Weight fragment buffers (ushort element offsets inside d_ws).
// One fragment = 32 lanes * 16 bf16 = 512 elements (1 KB).
// Layer 0 : Kc=1, Nt=16 ->   16*512 elems
// Layers 1..7 (layer 5 = W5h part): Kc=8, Nt=16 -> 128*512 elems each
// W5x (skip K-chunk): Kc=1, Nt=16 ; Wout: Kc=8, Nt=1
__host__ __device__ constexpr size_t woff(int l) {
    return (l == 0) ? (size_t)0 : (size_t)(16 * 512) + (size_t)(l - 1) * (8 * 16 * 512);
}
constexpr size_t kW5xOff  = (size_t)(16 * 512) + (size_t)7 * (8 * 16 * 512);  // 466944
constexpr size_t kWoutOff = kW5xOff + (size_t)(16 * 512);                     // 475136

__device__ __host__ inline unsigned short f2bf(float f) {
    union { float f; unsigned int u; } c;
    c.f = f;
    unsigned int u = c.u;
    u += 0x7FFFu + ((u >> 16) & 1u);      // round-to-nearest-even
    return (unsigned short)(u >> 16);
}

// ---------------------------------------------------------------------------
// Weight swizzle: fp32 (K x N row-major, leading dim src_ld) -> bf16 fragments.
// B-fragment lane layout (16-bit B 32x16): column = lane%16; lanes 0-15 hold
// K = kc*32 + {0..7,16..23}, lanes 16-31 hold K = kc*32 + {8..15,24..31}, 2 values per dword.
// ---------------------------------------------------------------------------
__global__ void swizzle_w(const float* __restrict__ src, unsigned short* __restrict__ dst,
                          int Ksrc, int row_off, int Nsrc, int Kc, int Nt, int src_ld) {
    int t = blockIdx.x * blockDim.x + threadIdx.x;
    if (t >= Kc * Nt * 32) return;
    int lane  = t & 31;
    int frag  = t >> 5;
    int nt    = frag % Nt;
    int kc    = frag / Nt;
    int khalf = lane >> 4;
    int n     = nt * 16 + (lane & 15);
    unsigned short* d = dst + (size_t)t * 16;
    for (int j = 0; j < 8; ++j) {
        for (int h = 0; h < 2; ++h) {
            int k = kc * 32 + ((j < 4) ? (khalf * 8 + j * 2 + h) : (16 + khalf * 8 + (j - 4) * 2 + h));
            float f = (k < Ksrc && n < Nsrc) ? src[(size_t)(row_off + k) * src_ld + n] : 0.0f;
            d[j * 2 + h] = f2bf(f);
        }
    }
}

__global__ void zero_f4(float* __restrict__ p, int n4) {
    int i = blockIdx.x * blockDim.x + threadIdx.x;
    if (i < n4) {
        u32x4 z; z[0] = 0u; z[1] = 0u; z[2] = 0u; z[3] = 0u;
        ((u32x4*)p)[i] = z;
    }
}

// ---------------------------------------------------------------------------
// Fragment loads (16B-aligned b128 accesses).
// A (16-bit 16x32, ISA layout): lane row = r0+lane%16; lanes 0-15 hold
// K = {0..7,16..23}, lanes 16-31 hold K = {8..15,24..31} -> two b128 loads.
// ---------------------------------------------------------------------------
union V16U { u32x4 u[2]; v16bf v; };

__device__ inline v16bf load_a_frag(const unsigned short* A, int stride, int row, int koff) {
    const unsigned short* p = A + (size_t)row * stride + koff;
    V16U t;
    t.u[0] = *(const u32x4*)(p);
    t.u[1] = *(const u32x4*)(p + 16);   // +32 bytes -> K+16 group
    return t.v;
}

__device__ inline v16bf load_b_frag(const unsigned short* W, int frag, int lane) {
    const unsigned short* p = W + ((size_t)frag * 32 + lane) * 16;
    V16U t;
    t.u[0] = *(const u32x4*)(p);
    t.u[1] = *(const u32x4*)(p + 8);
    return t.v;
}

__device__ inline v8f wmma_bf16(v16bf a, v16bf b, v8f c) {
    return __builtin_amdgcn_wmma_f32_16x16x32_bf16(false, a, false, b, (short)0, c,
                                                   false, false);
}

// ---------------------------------------------------------------------------
// Fused MLP: each block = 128 rows; 16 waves; wave (w&7) owns rows (w&7)*16,
// wave (w>>3) picks column half. Activations ping-pong in LDS as bf16.
// ---------------------------------------------------------------------------
__global__ __launch_bounds__(512, 1) void fern_mlp(
    const float* __restrict__ y,
    const unsigned short* __restrict__ WB,
    const float* __restrict__ b0, const float* __restrict__ b1,
    const float* __restrict__ b2, const float* __restrict__ b3,
    const float* __restrict__ b4, const float* __restrict__ b5,
    const float* __restrict__ b6, const float* __restrict__ b7,
    const float* __restrict__ bout,
    float* __restrict__ out) {
    extern __shared__ unsigned short smem[];
    unsigned short* act0 = smem;                      // 128 x 264 bf16
    unsigned short* act1 = act0 + M_TILE * LDS_T;     // 128 x 264 bf16
    unsigned short* xb   = act1 + M_TILE * LDS_T;     // 128 x 32  bf16 (patches)

    const int tid   = threadIdx.x;
    const int lane  = tid & 31;
    const int wave  = tid >> 5;
    const int rlane = lane & 15;
    const int khalf = lane >> 4;
    const int r0    = (wave & 7) * 16;
    const int chalf = wave >> 3;
    const size_t mbase = (size_t)blockIdx.x * M_TILE;

    // Gather 3x3 patches into LDS, padded K=9 -> 32 with zeros (bf16).
    for (int idx = tid; idx < M_TILE * 32; idx += 512) {
        int row = idx >> 5, c = idx & 31;
        size_t rg = mbase + (size_t)row;
        unsigned short v = 0;
        if (rg < (size_t)M_ROWS && c < 9) {
            int b  = (int)(rg & 63);
            int ij = (int)(rg >> 6);
            int pj = ij % NPS;
            int pi = ij / NPS;
            v = f2bf(y[(size_t)b * (IMS * IMS) + (size_t)(pi + c / 3) * IMS + (pj + c % 3)]);
        }
        xb[idx] = v;
    }
    __syncthreads();

    const float* bias[8] = {b0, b1, b2, b3, b4, b5, b6, b7};
    unsigned short* cur = act0;
    unsigned short* nxt = act1;

    #pragma unroll 1
    for (int l = 0; l < 8; ++l) {
        const unsigned short* A  = (l == 0) ? xb : cur;
        const int astride        = (l == 0) ? 32 : LDS_T;
        const int Kc             = (l == 0) ? 1 : 8;
        const unsigned short* Wf = WB + woff(l);

        if (l < 7) {  // stage next layer's fragments toward L2/L0
            __builtin_prefetch((const char*)(WB + woff(l + 1)) + (size_t)tid * 256, 0, 1);
        }

        v8f acc[8];
        for (int q = 0; q < 8; ++q)
            for (int e = 0; e < 8; ++e) acc[q][e] = 0.0f;

        for (int kc = 0; kc < Kc; ++kc) {
            v16bf a = load_a_frag(A, astride, r0 + rlane, kc * 32 + khalf * 8);
            #pragma unroll
            for (int q = 0; q < 8; ++q) {
                int nt = chalf * 8 + q;
                v16bf b = load_b_frag(Wf, kc * 16 + nt, lane);
                acc[q] = wmma_bf16(a, b, acc[q]);
            }
        }
        if (l == 5) {
            // concat([x, h]) skip: extra K-chunk x @ W5x (first 9 rows of w5)
            const unsigned short* Wx = WB + kW5xOff;
            v16bf a = load_a_frag(xb, 32, r0 + rlane, khalf * 8);
            #pragma unroll
            for (int q = 0; q < 8; ++q) {
                int nt = chalf * 8 + q;
                v16bf b = load_b_frag(Wx, nt, lane);
                acc[q] = wmma_bf16(a, b, acc[q]);
            }
        }

        // bias + ReLU, store bf16 activations (C layout: vgpr i -> row khalf*8+i,
        // col = lane%16 within the 16-wide tile).
        #pragma unroll
        for (int q = 0; q < 8; ++q) {
            int col  = (chalf * 8 + q) * 16 + rlane;
            float bb = bias[l][col];
            #pragma unroll
            for (int i = 0; i < 8; ++i) {
                float v = acc[q][i] + bb;
                v = v > 0.0f ? v : 0.0f;
                int row = r0 + khalf * 8 + i;
                nxt[(size_t)row * LDS_T + col] = f2bf(v);
            }
        }
        __syncthreads();
        unsigned short* tsw = cur; cur = nxt; nxt = tsw;
    }

    // Output layer 256 -> 9 (padded to 16 cols) + overlap-add scatter.
    if (chalf == 0) {
        v8f acc;
        for (int e = 0; e < 8; ++e) acc[e] = 0.0f;
        const unsigned short* Wo = WB + kWoutOff;
        #pragma unroll
        for (int kc = 0; kc < 8; ++kc) {
            v16bf a = load_a_frag(cur, LDS_T, r0 + rlane, kc * 32 + khalf * 8);
            v16bf b = load_b_frag(Wo, kc, lane);
            acc = wmma_bf16(a, b, acc);
        }
        int c = rlane;
        if (c < 9) {
            float bb = bout[c];
            int pr = c / 3, pc = c % 3;
            #pragma unroll
            for (int i = 0; i < 8; ++i) {
                size_t rg = mbase + (size_t)(r0 + khalf * 8 + i);
                if (rg < (size_t)M_ROWS) {
                    int b  = (int)(rg & 63);
                    int ij = (int)(rg >> 6);
                    int pj = ij % NPS;
                    int pi = ij / NPS;
                    atomicAdd(out + (size_t)b * (IMS * IMS)
                                  + (size_t)(pi + pr) * IMS + (pj + pc),
                              acc[i] + bb);
                }
            }
        }
    }
}

// ---------------------------------------------------------------------------
extern "C" void kernel_launch(void* const* d_in, const int* in_sizes, int n_in,
                              void* d_out, int out_size, void* d_ws, size_t ws_size,
                              hipStream_t stream) {
    (void)in_sizes; (void)n_in; (void)ws_size;
    const float* y = (const float*)d_in[0];
    const float* w[8]; const float* bl[8];
    for (int l = 0; l < 8; ++l) {
        w[l]  = (const float*)d_in[1 + 2 * l];
        bl[l] = (const float*)d_in[2 + 2 * l];
    }
    const float* wout = (const float*)d_in[17];
    const float* bout = (const float*)d_in[18];
    float* out = (float*)d_out;
    unsigned short* WB = (unsigned short*)d_ws;   // ~0.96 MB bf16 fragments

    // d_out is poisoned: zero before scatter-add.
    {
        int n4 = out_size / 4;
        zero_f4<<<(n4 + 255) / 256, 256, 0, stream>>>(out, n4);
    }

    auto swz = [&](const float* s, size_t off, int Ksrc, int ro, int Nsrc,
                   int Kc, int Nt, int ld) {
        int tot = Kc * Nt * 32;
        swizzle_w<<<(tot + 255) / 256, 256, 0, stream>>>(s, WB + off, Ksrc, ro, Nsrc,
                                                         Kc, Nt, ld);
    };
    swz(w[0], woff(0), 9, 0, 256, 1, 16, 256);          // layer 0: 9(->32) x 256
    for (int l = 1; l < 8; ++l) {
        if (l == 5) swz(w[5], woff(5), 256, 9, 256, 8, 16, 256);  // W5h: rows 9..264
        else        swz(w[l], woff(l), 256, 0, 256, 8, 16, 256);
    }
    swz(w[5], kW5xOff, 9, 0, 256, 1, 16, 256);          // W5x: rows 0..8 of w5
    swz(wout, kWoutOff, 256, 0, 9, 8, 1, 9);            // 256 x 9(->16)

    size_t smem = (size_t)(2 * M_TILE * LDS_T + M_TILE * 32) * sizeof(unsigned short);
    fern_mlp<<<NUM_TILES, 512, smem, stream>>>(y, WB,
                                               bl[0], bl[1], bl[2], bl[3],
                                               bl[4], bl[5], bl[6], bl[7],
                                               bout, out);
}
